// TT_75849122447380
// MI455X (gfx1250) — matrix operand order repaired and checked by reading, not compiled, not run on hardware
//
#include <hip/hip_runtime.h>
#include <hip/hip_bf16.h>
#include <math.h>

typedef __attribute__((ext_vector_type(16))) _Float16 v16h;
typedef __attribute__((ext_vector_type(8)))  _Float16 v8h;
typedef __attribute__((ext_vector_type(8)))  float    v8f;

#define NPOS   784
#define BATCH  256
#define NSEG   8
#define SEGLEN (NPOS / NSEG)   // 98, exact; 97 fold steps per segment

#define LOG2E_F 1.4426950408889634f
#define LN2_F   0.6931471805599453f

// Raw hardware transcendentals: single v_exp_f32 / v_log_f32 / v_rcp_f32.
static __device__ __forceinline__ float fast_exp2(float x) {
  return __builtin_amdgcn_exp2f(x);
}
static __device__ __forceinline__ float fast_log2(float x) {
  return __builtin_amdgcn_logf(x);
}
static __device__ __forceinline__ float fast_rcp(float x) {
  return __builtin_amdgcn_rcpf(x);
}

// Value of this register at lane (i ^ 16): v_permlanex16_b32 with identity
// nibble selects (pure VALU, no DS pipe) if available, else generic shuffle.
static __device__ __forceinline__ float xswap16(float v) {
#if __has_builtin(__builtin_amdgcn_permlanex16)
  return __int_as_float(__builtin_amdgcn_permlanex16(
      __float_as_int(v), __float_as_int(v), 0x76543210u, 0xfedcba98u,
      false, false));
#else
  return __shfl_xor(v, 16, 32);
#endif
}

// Cross-half pair exchange with a SINGLE cross-lane op:
//   result a' = {a.lanes0-15, b.lanes0-15}, b' = {a.lanes16-31, b.lanes16-31}.
// Lower lanes only need partner's a; upper lanes only need partner's b, so
// route t = hi ? a : b through one swap.
static __device__ __forceinline__ void half_swap(float& a, float& b, bool hi) {
  const float t = hi ? a : b;
  const float q = xswap16(t);
  a = hi ? q : a;
  b = hi ? b : q;
}

// max with the partner lane in the other 16-lane half
static __device__ __forceinline__ float max_xhalf(float c) {
  return fmaxf(c, xswap16(c));
}

static __device__ __forceinline__ float softplus_f(float t) {
  return (t > 0.f) ? (t + log1pf(__expf(-t))) : log1pf(__expf(t));
}

static __device__ __forceinline__ v16h mk_v16h(v8h lo, v8h hi) {
  v16h r;
#pragma unroll
  for (int t = 0; t < 8; ++t) { r[t] = lo[t]; r[t + 8] = hi[t]; }
  return r;
}

static __device__ __forceinline__ v8f wmma_f32_f16(v16h a, v16h b) {
  v8f c = {};
  return __builtin_amdgcn_wmma_f32_16x16x32_f16(false, a, false, b, (short)0, c,
                                                false, false);
}

// ---------------------------------------------------------------------------
// Pre-pass 1 (cold): per position i, log_softmax (log2 units) and per-ROW
// maxes of the two possible joint matrices M2_i[x] (log2 units).
// ---------------------------------------------------------------------------
__global__ void __launch_bounds__(32) prep1_kernel(
    const float* __restrict__ ulpa, const float* __restrict__ theta,
    float* __restrict__ logp2, float* __restrict__ rmax2) {
  const int i = blockIdx.x;
  const int lane = threadIdx.x;
  const float* th = theta + (size_t)i * 1024;

  // lane = column n for the softmax part
  float u = ulpa[i * 32 + lane];
  float mx = u;
#pragma unroll
  for (int m = 1; m < 32; m <<= 1) mx = fmaxf(mx, __shfl_xor(mx, m, 32));
  float s = __expf(u - mx);
#pragma unroll
  for (int m = 1; m < 32; m <<= 1) s += __shfl_xor(s, m, 32);
  const float lp2 = (u - mx - __logf(s)) * LOG2E_F;
  logp2[i * 32 + lane] = lp2;

  // lane = row r for the rowmax part; lp2[n] broadcast from lane n
  float rm0 = -1e30f, rm1 = -1e30f;
#pragma unroll 4
  for (int n = 0; n < 32; ++n) {
    const float t = th[lane * 32 + n];
    const float lpn = __shfl(lp2, n, 32);
    const float m0 = lpn - softplus_f(t) * LOG2E_F;  // x=0 joint (log2)
    const float m1 = m0 + t * LOG2E_F;               // x=1
    rm0 = fmaxf(rm0, m0);
    rm1 = fmaxf(rm1, m1);
  }
  rmax2[(i * 2 + 0) * 32 + lane] = rm0;
  rmax2[(i * 2 + 1) * 32 + lane] = rm1;
}

// ---------------------------------------------------------------------------
// Pre-pass 2 (cold): chained A tables for the right-to-left fold. For
// i in 0..782 and bit pair (xn = x_{i+1}, xc = x_i):
//   Atil_i(m,k) = 2^( M2_i(m,k)[xc] + rmax2[i+1][xn](k) - rmax2[i][xc](m) )
// Stored row-major (m,k) f16 so WMMA A fragments load as plain 16B vectors.
// All entries <= 1 (f16-safe).
// ---------------------------------------------------------------------------
__global__ void __launch_bounds__(32) prep2_kernel(
    const float* __restrict__ theta, const float* __restrict__ logp2,
    const float* __restrict__ rmax2, _Float16* __restrict__ aexpC) {
  const int i = blockIdx.x;       // 0..782
  const int m = threadIdx.x;      // row
  const float* th = theta + (size_t)i * 1024;

  const float ra0 = rmax2[(i * 2 + 0) * 32 + m];
  const float ra1 = rmax2[(i * 2 + 1) * 32 + m];

  _Float16* o00 = aexpC + (size_t)(i * 4 + 0) * 1024 + m * 32;  // xn=0,xc=0
  _Float16* o01 = aexpC + (size_t)(i * 4 + 1) * 1024 + m * 32;  // xn=0,xc=1
  _Float16* o10 = aexpC + (size_t)(i * 4 + 2) * 1024 + m * 32;  // xn=1,xc=0
  _Float16* o11 = aexpC + (size_t)(i * 4 + 3) * 1024 + m * 32;  // xn=1,xc=1

#pragma unroll 4
  for (int k = 0; k < 32; ++k) {
    const float t = th[m * 32 + k];
    const float lpk = logp2[i * 32 + k];
    const float b0 = lpk - softplus_f(t) * LOG2E_F;  // M2_i(m,k), x=0
    const float b1 = b0 + t * LOG2E_F;               // x=1
    const float gn0 = rmax2[((i + 1) * 2 + 0) * 32 + k];
    const float gn1 = rmax2[((i + 1) * 2 + 1) * 32 + k];
    o00[k] = (_Float16)fast_exp2(b0 + gn0 - ra0);
    o01[k] = (_Float16)fast_exp2(b1 + gn0 - ra1);
    o10[k] = (_Float16)fast_exp2(b0 + gn1 - ra0);
    o11[k] = (_Float16)fast_exp2(b1 + gn1 - ra1);
  }
}

// ---------------------------------------------------------------------------
// One right-fold chain step: L <- Atil_i @ L. The running product is the B
// operand; C-layout -> B-layout uses one cross-half op per register pair.
// NORM: per-COLUMN normalize (lane-local: 2 rcp + 2 log2 per lane). Skipping
// alternate norms is safe: one step grows colmax by <= 32x << f16 max 65504.
// ---------------------------------------------------------------------------
template <bool NORM>
static __device__ __forceinline__ void chain_step(
    float L[2][2][8], float ca[2], const _Float16* __restrict__ ap,
    int half, int cin) {
  const bool hi = half != 0;

  // A fragments from the precomputed row-major table
  v16h a[2];
#pragma unroll
  for (int mt = 0; mt < 2; ++mt) {
    const int base = (16 * mt + cin) * 32 + half * 16;
    a[mt] = mk_v16h(*(const v8h*)&ap[base], *(const v8h*)&ap[base + 8]);
  }

  float r[2] = {1.f, 1.f};
  if (NORM) {
#pragma unroll
    for (int nt = 0; nt < 2; ++nt) {
      float c = L[0][nt][0];
#pragma unroll
      for (int v = 1; v < 8; ++v) c = fmaxf(c, L[0][nt][v]);
#pragma unroll
      for (int v = 0; v < 8; ++v) c = fmaxf(c, L[1][nt][v]);
      c = max_xhalf(c);
      r[nt] = fast_rcp(c);
      ca[nt] += fast_log2(c);
    }
  }

  // B fragments: one cross-half op per (nt, v)
  v16h bb[2];
#pragma unroll
  for (int nt = 0; nt < 2; ++nt) {
#pragma unroll
    for (int v = 0; v < 8; ++v) {
      float p0 = NORM ? L[0][nt][v] * r[nt] : L[0][nt][v];  // rows v | 8+v
      float p1 = NORM ? L[1][nt][v] * r[nt] : L[1][nt][v];  // rows 16+v | 24+v
      half_swap(p0, p1, hi);  // p0: rows v|16+v (K 0..7|16..23); p1: 8+v|24+v
      bb[nt][v]     = (_Float16)p0;
      bb[nt][8 + v] = (_Float16)p1;
    }
  }

#pragma unroll
  for (int mt = 0; mt < 2; ++mt)
#pragma unroll
    for (int nt = 0; nt < 2; ++nt) {
      const v8f d = wmma_f32_f16(a[mt], bb[nt]);
#pragma unroll
      for (int v = 0; v < 8; ++v) L[mt][nt][v] = d[v];
    }
}

// ---------------------------------------------------------------------------
// Segment chain (right-to-left): one wave folds 98 positions for (b, s).
// State: L (linear f32) + ca (per-column log2 scale). Row scale is carried
// implicitly by the chained A tables; applied once at segment end.
// No LDS, no barriers in the hot loop.
// ---------------------------------------------------------------------------
__global__ void __launch_bounds__(32) seg_kernel(
    const float* __restrict__ x, const float* __restrict__ theta,
    const float* __restrict__ logp2, const float* __restrict__ rmax2,
    const _Float16* __restrict__ aexpC, float* __restrict__ part) {
  const int b = blockIdx.x, s = blockIdx.y;
  const int lane = threadIdx.x;
  const int half = lane >> 4, cin = lane & 15;
  const int i0 = s * SEGLEN;
  const int ilast = i0 + SEGLEN - 1;

  float L[2][2][8];
  float ca[2] = {0.f, 0.f};
  int xn;
  {  // init at rightmost position: L = 2^(M2_ilast - rowmax), row-normalized
    const float xb = x[b * NPOS + ilast];
    xn = (xb > 0.5f) ? 1 : 0;
    const float* th = theta + (size_t)ilast * 1024;
    const float* lp = logp2 + ilast * 32;
    const float* rmx = rmax2 + (size_t)(ilast * 2 + xn) * 32;
#pragma unroll
    for (int mt = 0; mt < 2; ++mt)
#pragma unroll
      for (int nt = 0; nt < 2; ++nt)
#pragma unroll
        for (int v = 0; v < 8; ++v) {
          const int row = 16 * mt + 8 * half + v;
          const int col = 16 * nt + cin;
          const float t = th[row * 32 + col];
          const float p2 = (xb * t - softplus_f(t)) * LOG2E_F + lp[col];
          L[mt][nt][v] = fast_exp2(p2 - rmx[row]);
        }
  }

  int i = ilast - 1;
  {  // first step (normalized); 96 remaining steps in skip/norm pairs
    const int xc = (x[b * NPOS + i] > 0.5f) ? 1 : 0;
    chain_step<true>(L, ca, aexpC + (size_t)(i * 4 + xn * 2 + xc) * 1024,
                     half, cin);
    xn = xc;
    --i;
  }
  for (int j = 0; j < (SEGLEN - 2) / 2; ++j) {
    int xc = (x[b * NPOS + i] > 0.5f) ? 1 : 0;
    chain_step<false>(L, ca, aexpC + (size_t)(i * 4 + xn * 2 + xc) * 1024,
                      half, cin);
    xn = xc;
    --i;
    xc = (x[b * NPOS + i] > 0.5f) ? 1 : 0;
    chain_step<true>(L, ca, aexpC + (size_t)(i * 4 + xn * 2 + xc) * 1024,
                     half, cin);
    xn = xc;
    --i;
  }
  // processed i0..ilast; xn == bit(i0)

  // back to log2 space: P = log2(L) + rowmax_{i0}(row) + ca(col)
  const float* rmx = rmax2 + (size_t)(i0 * 2 + xn) * 32;
  float rv[2][8];
#pragma unroll
  for (int mt = 0; mt < 2; ++mt)
#pragma unroll
    for (int v = 0; v < 8; ++v) rv[mt][v] = rmx[16 * mt + 8 * half + v];

  float* pp = part + ((size_t)(b * NSEG + s) * 32 + lane) * 32;
#pragma unroll
  for (int mt = 0; mt < 2; ++mt)
#pragma unroll
    for (int nt = 0; nt < 2; ++nt)
#pragma unroll
      for (int v = 0; v < 8; ++v)
        pp[mt * 16 + nt * 8 + v] =
            fast_log2(L[mt][nt][v]) + rv[mt][v] + ca[nt];
}

// ---------------------------------------------------------------------------
// Combine (cold, 7 steps): fold the 8 segment partials per batch element in
// log2 space, then diagonal log-sum-exp, scaled to natural log at the end.
// ---------------------------------------------------------------------------
__global__ void __launch_bounds__(32) comb_kernel(
    const float* __restrict__ part, float* __restrict__ out) {
  __shared__ __align__(16) _Float16 smA[1024];
  __shared__ __align__(16) _Float16 smB[1024];
  const int b = blockIdx.x;
  const int lane = threadIdx.x;
  const int half = lane >> 4, cin = lane & 15;

  float P[2][2][8];
  {
    const float* pp = part + ((size_t)(b * NSEG) * 32 + lane) * 32;
#pragma unroll
    for (int mt = 0; mt < 2; ++mt)
#pragma unroll
      for (int nt = 0; nt < 2; ++nt)
#pragma unroll
        for (int v = 0; v < 8; ++v) P[mt][nt][v] = pp[mt * 16 + nt * 8 + v];
  }

  for (int s = 1; s < NSEG; ++s) {
    const float* rp = part + ((size_t)(b * NSEG + s) * 32 + lane) * 32;
    float R[2][2][8];
#pragma unroll
    for (int mt = 0; mt < 2; ++mt)
#pragma unroll
      for (int nt = 0; nt < 2; ++nt)
#pragma unroll
        for (int v = 0; v < 8; ++v) R[mt][nt][v] = rp[mt * 16 + nt * 8 + v];

    float cm[2];
#pragma unroll
    for (int nt = 0; nt < 2; ++nt) {
      float c = -1e30f;
#pragma unroll
      for (int mt = 0; mt < 2; ++mt)
#pragma unroll
        for (int v = 0; v < 8; ++v) c = fmaxf(c, R[mt][nt][v]);
      c = max_xhalf(c);
      cm[nt] = c;
    }

    float rm[2][8];
#pragma unroll
    for (int mt = 0; mt < 2; ++mt)
#pragma unroll
      for (int v = 0; v < 8; ++v) {
        float t = fmaxf(P[mt][0][v], P[mt][1][v]);
#pragma unroll
        for (int m = 1; m < 16; m <<= 1) t = fmaxf(t, __shfl_xor(t, m, 32));
        rm[mt][v] = t;
      }

#pragma unroll
    for (int mt = 0; mt < 2; ++mt)
#pragma unroll
      for (int nt = 0; nt < 2; ++nt)
#pragma unroll
        for (int v = 0; v < 8; ++v) {
          const int row = 16 * mt + 8 * half + v;
          const int col = 16 * nt + cin;
          smA[row * 32 + col] = (_Float16)fast_exp2(P[mt][nt][v] - rm[mt][v]);
          smB[col * 32 + row] = (_Float16)fast_exp2(R[mt][nt][v] - cm[nt]);
        }
    __syncthreads();

    v16h a[2], bb[2];
#pragma unroll
    for (int mt = 0; mt < 2; ++mt) {
      const int base = (16 * mt + cin) * 32 + half * 16;
      a[mt] = mk_v16h(*(const v8h*)&smA[base], *(const v8h*)&smA[base + 8]);
    }
#pragma unroll
    for (int nt = 0; nt < 2; ++nt) {
      const int off = (16 * nt + cin) * 32 + half * 16;
      bb[nt] = mk_v16h(*(const v8h*)&smB[off], *(const v8h*)&smB[off + 8]);
    }

#pragma unroll
    for (int mt = 0; mt < 2; ++mt)
#pragma unroll
      for (int nt = 0; nt < 2; ++nt) {
        const v8f d = wmma_f32_f16(a[mt], bb[nt]);
#pragma unroll
        for (int v = 0; v < 8; ++v)
          P[mt][nt][v] = fast_log2(d[v]) + rm[mt][v] + cm[nt];
      }
    __syncthreads();
  }

  float d0 = -1e30f, d1 = -1e30f;
  if (lane < 8) {
    d0 = P[0][0][lane];
    d1 = P[1][1][lane];
  } else if (lane >= 24) {
    d0 = P[0][0][lane - 24];
    d1 = P[1][1][lane - 24];
  }
  float mx = fmaxf(d0, d1);
#pragma unroll
  for (int m = 1; m < 32; m <<= 1) mx = fmaxf(mx, __shfl_xor(mx, m, 32));
  float ss = fast_exp2(d0 - mx) + fast_exp2(d1 - mx);
#pragma unroll
  for (int m = 1; m < 32; m <<= 1) ss += __shfl_xor(ss, m, 32);
  if (lane == 0) out[b] = (mx + fast_log2(ss)) * LN2_F;
}

extern "C" void kernel_launch(void* const* d_in, const int* in_sizes, int n_in,
                              void* d_out, int out_size, void* d_ws,
                              size_t ws_size, hipStream_t stream) {
  (void)in_sizes; (void)n_in; (void)out_size; (void)ws_size;
  const float* x     = (const float*)d_in[0];
  const float* ulpa  = (const float*)d_in[1];
  const float* theta = (const float*)d_in[2];
  float* out = (float*)d_out;

  char* ws = (char*)d_ws;
  float*    logp2 = (float*)ws;                            // 784*32 f32
  float*    rmax2 = (float*)(ws + (size_t)NPOS * 32 * 4);  // 784*2*32 f32
  _Float16* aexpC = (_Float16*)(ws + (size_t)NPOS * 32 * 4 +
                                (size_t)NPOS * 64 * 4);    // 784*4*1024 f16
  float*    part  = (float*)(ws + (size_t)NPOS * 32 * 4 +
                             (size_t)NPOS * 64 * 4 +
                             (size_t)NPOS * 4096 * 2);     // 256*8*1024 f32

  prep1_kernel<<<NPOS, 32, 0, stream>>>(ulpa, theta, logp2, rmax2);
  prep2_kernel<<<NPOS - 1, 32, 0, stream>>>(theta, logp2, rmax2, aexpC);
  seg_kernel<<<dim3(BATCH, NSEG), 32, 0, stream>>>(x, theta, logp2, rmax2,
                                                   aexpC, part);
  comb_kernel<<<BATCH, 32, 0, stream>>>(part, out);

  // second output of the tuple: theta passthrough
  hipMemcpyAsync(out + BATCH, theta, (size_t)NPOS * 1024 * sizeof(float),
                 hipMemcpyDeviceToDevice, stream);
}
